// GPT_83004537962572
// MI455X (gfx1250) — matrix operand order repaired and checked by reading, not compiled, not run on hardware
//
#include <hip/hip_runtime.h>
#include <hip/hip_bf16.h>
#include <math.h>

// ---------------- constants ----------------
constexpr int NL   = 12;
constexpr int DMOD = 768;
constexpr int NH   = 12;
constexpr int HSZ  = 64;
constexpr int TS   = 1024;
constexpr int NB   = 4;
constexpr int MT   = NB * TS;     // 4096 tokens
constexpr int NV   = 50257;

typedef unsigned short u16;
typedef unsigned int   u32;
typedef unsigned long long u64;

typedef __attribute__((ext_vector_type(16))) __bf16 v16bf;
typedef __attribute__((ext_vector_type(8)))  float  v8f;
typedef __attribute__((ext_vector_type(4)))  u32    v4u;
typedef __attribute__((ext_vector_type(8)))  int    v8i;
typedef __attribute__((ext_vector_type(4)))  int    v4i;

union FragA { uint4 q[2]; v16bf v; };

template <bool B_> struct BoolC { static constexpr bool value = B_; };

__device__ __forceinline__ u16 f2bf(float f) {
  u32 u = __float_as_uint(f);
  u32 r = u + 0x7FFFu + ((u >> 16) & 1u);   // round-to-nearest-even
  return (u16)(r >> 16);
}
__device__ __forceinline__ float bf2f(u16 h) { return __uint_as_float(((u32)h) << 16); }
__device__ __forceinline__ v8f vzero() {
  v8f z = {0.f,0.f,0.f,0.f,0.f,0.f,0.f,0.f};
  return z;
}

// ds_swizzle_b32 XOR-lane shuffle: offset = (xor_mask<<10) | 0x1F
template <int IMM>
__device__ __forceinline__ float swz(float x) {
  return __uint_as_float((u32)__builtin_amdgcn_ds_swizzle((int)__float_as_uint(x), IMM));
}
__device__ __forceinline__ float half16_max(float x) {
  x = fmaxf(x, swz<0x041F>(x));
  x = fmaxf(x, swz<0x081F>(x));
  x = fmaxf(x, swz<0x101F>(x));
  x = fmaxf(x, swz<0x201F>(x));
  return x;
}
__device__ __forceinline__ float half16_sum(float x) {
  x += swz<0x041F>(x);
  x += swz<0x081F>(x);
  x += swz<0x101F>(x);
  x += swz<0x201F>(x);
  return x;
}
__device__ __forceinline__ float wave32_sum(float x) {
  x += swz<0x041F>(x);
  x += swz<0x081F>(x);
  x += swz<0x101F>(x);
  x += swz<0x201F>(x);
  x += swz<0x401F>(x);
  return x;
}

// ---- Tensor Data Mover: 2D bf16 tile -> LDS with row padding (ISA cdna5 §8) ----
// D# group0: count=1, lds_addr, global_addr, type=2
// D# group1: data_size=1(2B), pad fields, tensor dims/strides, tile dims
// amdgpu-toolchain (clang-23) 6-arg builtin form:
//   (uint32x4 g0, int32x8 g1, int32x4 g2, int32x4 g3, int32x8, i32 cpol)
__device__ __forceinline__ void tdm_load_2d(u32 lds_off, const void* gptr,
                                            u32 stride_elems, u32 tile_d0, u32 tile_d1,
                                            u32 pad_interval, u32 pad_amount,
                                            u32 tensor_d0, u32 tensor_d1) {
  u64 ga = (u64)(size_t)gptr;
  v4u g0;
  g0[0] = 1u;                                       // count=1, user mode
  g0[1] = lds_off;                                  // lds_addr [63:32]
  g0[2] = (u32)ga;                                  // global_addr low
  g0[3] = (u32)(ga >> 32) | (2u << 30);             // global_addr high | type=2
  u64 b0 = (1ull << 16)                             // data_size = 2 bytes
         | (1ull << 20)                             // pad_enable
         | ((u64)pad_interval << 22)
         | ((u64)pad_amount << 25)
         | ((u64)(tensor_d0 & 0xFFFFu) << 48);      // tensor_dim0[15:0]
  u64 b1 = ((u64)tensor_d0 >> 16)                   // tensor_dim0[31:16]
         | ((u64)tensor_d1 << 16)                   // tensor_dim1
         | ((u64)tile_d0 << 48);                    // tile_dim0
  u64 b2 = ((u64)tile_d1)                           // tile_dim1 (tile_dim2=0)
         | ((u64)stride_elems << 32);               // tensor_dim0_stride[31:0]
  u64 b3 = 0;                                       // stride[47:32]=0, dim1_stride=0
  v8i g1;
  g1[0] = (int)(u32)b0; g1[1] = (int)(u32)(b0 >> 32);
  g1[2] = (int)(u32)b1; g1[3] = (int)(u32)(b1 >> 32);
  g1[4] = (int)(u32)b2; g1[5] = (int)(u32)(b2 >> 32);
  g1[6] = (int)(u32)b3; g1[7] = (int)(u32)(b3 >> 32);
  v4i z4 = {0, 0, 0, 0};
  v8i z8 = {0, 0, 0, 0, 0, 0, 0, 0};
  __builtin_amdgcn_tensor_load_to_lds(g0, g1, z4, z4, z8, 0);
}

// ---------------- embedding: R = wte[idx] + wpe ----------------
__global__ void embed_kernel(const int* __restrict__ idx, const float* __restrict__ wte,
                             const float* __restrict__ wpe, float* __restrict__ R) {
  int i   = blockIdx.x * blockDim.x + threadIdx.x;   // over MT*DMOD
  int tok = i / DMOD, d = i % DMOD;
  int t   = tok % TS;
  R[i] = wte[(size_t)idx[tok] * DMOD + d] + wpe[(size_t)t * DMOD + d];
}

// ---------------- layernorm (fp32 in, bf16 out), one block per row ----------------
__global__ void ln_kernel(const float* __restrict__ X, const float* __restrict__ w,
                          const float* __restrict__ bvec, u16* __restrict__ O) {
  int row = blockIdx.x;
  const float* x = X + (size_t)row * DMOD;
  int tid = threadIdx.x;
  float v0 = x[tid], v1 = x[tid + 256], v2 = x[tid + 512];
  float s  = wave32_sum(v0 + v1 + v2);
  float s2 = wave32_sum(v0 * v0 + v1 * v1 + v2 * v2);
  __shared__ float red[2][8];
  __shared__ float stat[2];
  int lane = tid & 31, wid = tid >> 5;
  if (lane == 0) { red[0][wid] = s; red[1][wid] = s2; }
  __syncthreads();
  if (tid == 0) {
    float a = 0.f, b2 = 0.f;
    #pragma unroll
    for (int i = 0; i < 8; ++i) { a += red[0][i]; b2 += red[1][i]; }
    float mean = a / (float)DMOD;
    float var  = b2 / (float)DMOD - mean * mean;
    stat[0] = mean; stat[1] = rsqrtf(var + 1e-5f);
  }
  __syncthreads();
  float mean = stat[0], rstd = stat[1];
  u16* o = O + (size_t)row * DMOD;
  o[tid]       = f2bf((v0 - mean) * rstd * w[tid]       + bvec[tid]);
  o[tid + 256] = f2bf((v1 - mean) * rstd * w[tid + 256] + bvec[tid + 256]);
  o[tid + 512] = f2bf((v2 - mean) * rstd * w[tid + 512] + bvec[tid + 512]);
}

// ---------------- generic bf16 WMMA GEMM (TDM A-tiles, double-buffered) ----------------
// MODE 0: store bf16   MODE 1: gelu->bf16   MODE 2: resid_f32 += acc+bias   MODE 3: f32 store
// BT: B accessed transposed (wte rows). Block 128 threads (2x2 waves), wave tile 64x64.
template <int MODE, bool BT>
__global__ void __launch_bounds__(128) gemm_kernel(
    const u16* __restrict__ A, const float* __restrict__ B,
    const float* __restrict__ bias,
    float* __restrict__ Rout, u16* __restrict__ Obf,
    int M, int N, int K, int lda, int ldb) {
  __shared__ __align__(16) u16 smem[20480];                 // 40KB
  u16 (*As)[128][40] = (u16 (*)[128][40])smem;              // 2 x 10240B
  u16 (*Bs)[128][40] = (u16 (*)[128][40])(smem + 10240);    // 2 x 10240B
  const u32 ldsbase = (u32)(size_t)(void*)smem;

  const int tid   = threadIdx.x;
  const int lane  = tid & 31, wid = tid >> 5;
  const int uwid  = __builtin_amdgcn_readfirstlane(wid);    // scalar wave id
  const int hi    = lane >> 4;
  const int l15   = lane & 15;
  const int waveM = wid >> 1;
  const int waveN = wid & 1;
  const int rowBase = blockIdx.y * 128;
  const int colBase = blockIdx.x * 128;

  v8f acc[4][4];
  #pragma unroll
  for (int mi = 0; mi < 4; ++mi)
    #pragma unroll
    for (int ni = 0; ni < 4; ++ni) acc[mi][ni] = vzero();

  const int nk = K / 32;

  auto tdmA = [&](int buf, int k0) {
    // 128x32 bf16 tile; LDS rows 64B + 16B pad: pad_interval=3 (16 dwords), pad_amount=3 (4 dwords)
    tdm_load_2d(ldsbase + buf * 10240,
                A + (size_t)rowBase * lda + k0,
                (u32)lda, 32, 128, 3, 3, (u32)lda, (u32)M);
  };
  auto stageB = [&](int buf, int k0) {
    if (BT) {
      #pragma unroll
      for (int i = 0; i < 8; ++i) {
        int s = tid + i * 128;
        int r = s >> 3, c = (s & 7) * 4;
        int gn = colBase + r;
        float4 f = make_float4(0.f, 0.f, 0.f, 0.f);
        if (gn < N) f = *(const float4*)(B + (size_t)gn * ldb + k0 + c);
        u16 t4[4] = { f2bf(f.x), f2bf(f.y), f2bf(f.z), f2bf(f.w) };
        *(uint2*)&Bs[buf][r][c] = *(const uint2*)t4;
      }
    } else {
      const float* bp = B + (size_t)k0 * ldb + colBase + tid;
      #pragma unroll
      for (int g = 0; g < 8; ++g) {
        u16 t4[4];
        #pragma unroll
        for (int j = 0; j < 4; ++j) t4[j] = f2bf(bp[(size_t)(g * 4 + j) * ldb]);
        *(uint2*)&Bs[buf][tid][g * 4] = *(const uint2*)t4;
      }
    }
  };

  if (uwid == 0) tdmA(0, 0);
  stageB(0, 0);
  if (uwid == 0) __builtin_amdgcn_s_wait_tensorcnt(0);
  __syncthreads();

  for (int kt = 0; kt < nk; ++kt) {
    const int cur = kt & 1;
    if (kt + 1 < nk) {
      if (uwid == 0) tdmA(cur ^ 1, (kt + 1) * 32);
      stageB(cur ^ 1, (kt + 1) * 32);
    }
    FragA a[4], b[4];
    #pragma unroll
    for (int mi = 0; mi < 4; ++mi) {
      int r = waveM * 64 + mi * 16 + l15;
      a[mi].q[0] = *(const uint4*)&As[cur][r][hi * 8];
      a[mi].q[1] = *(const uint4*)&As[cur][r][hi * 8 + 16];
    }
    #pragma unroll
    for (int ni = 0; ni < 4; ++ni) {
      int c = waveN * 64 + ni * 16 + l15;
      b[ni].q[0] = *(const uint4*)&Bs[cur][c][hi * 8];
      b[ni].q[1] = *(const uint4*)&Bs[cur][c][hi * 8 + 16];
    }
    #pragma unroll
    for (int mi = 0; mi < 4; ++mi)
      #pragma unroll
      for (int ni = 0; ni < 4; ++ni)
        acc[mi][ni] = __builtin_amdgcn_wmma_f32_16x16x32_bf16(
            false, a[mi].v, false, b[ni].v, (short)0, acc[mi][ni], false, false);
    if (kt + 1 < nk && uwid == 0) __builtin_amdgcn_s_wait_tensorcnt(0);
    __syncthreads();
  }

  // ---- epilogue ----
  if (MODE == 0 || MODE == 1) {
    // scatter C tile (bias/gelu applied) into LDS, then coalesced b128 stores
    u16 (*Cs)[136] = (u16 (*)[136])smem;   // 128x136 u16 = 34816B (<= 40KB)
    #pragma unroll
    for (int mi = 0; mi < 4; ++mi) {
      #pragma unroll
      for (int ni = 0; ni < 4; ++ni) {
        int cc = waveN * 64 + ni * 16 + l15;
        float bv = bias[colBase + cc];
        #pragma unroll
        for (int r = 0; r < 8; ++r) {
          int rr = waveM * 64 + mi * 16 + r + hi * 8;
          float v = acc[mi][ni][r] + bv;
          if (MODE == 1)
            v = 0.5f * v * (1.f + tanhf(0.7978845608f * (v + 0.044715f * v * v * v)));
          Cs[rr][cc] = f2bf(v);
        }
      }
    }
    __syncthreads();
    #pragma unroll
    for (int i = 0; i < 16; ++i) {
      int s = tid + i * 128;
      int r = s >> 4, c = (s & 15) * 8;
      *(uint4*)(Obf + (size_t)(rowBase + r) * N + colBase + c) = *(const uint4*)&Cs[r][c];
    }
  } else {
    #pragma unroll
    for (int mi = 0; mi < 4; ++mi) {
      #pragma unroll
      for (int ni = 0; ni < 4; ++ni) {
        int col = colBase + waveN * 64 + ni * 16 + l15;
        float bv = (MODE == 2) ? bias[col] : 0.f;
        #pragma unroll
        for (int r = 0; r < 8; ++r) {
          int row = rowBase + waveM * 64 + mi * 16 + r + hi * 8;
          float v = acc[mi][ni][r] + bv;
          if (MODE == 2) {
            Rout[(size_t)row * N + col] += v;
          } else {
            if (col < N) Rout[(size_t)row * N + col] = v;
          }
        }
      }
    }
  }
}

// ---------------- flash attention: WMMA QK^T + PV (TDM K-tiles) ----------------
// grid (TS/64, NH, NB), block 128 (4 waves, each owns a 16-query tile)
__global__ void __launch_bounds__(128) attn_kernel(const u16* __restrict__ QKV,
                                                   u16* __restrict__ Y) {
  __shared__ __align__(16) u16 Kt[32][80];   // [key][feat], 160B rows
  __shared__ __align__(16) u16 Vt[64][40];   // [feat][key], 80B rows
  __shared__ __align__(16) u16 Pb[4][16][40];
  const int tid  = threadIdx.x;
  const int lane = tid & 31, wid = tid >> 5;
  const int uwid = __builtin_amdgcn_readfirstlane(wid);   // scalar wave id
  const int hi   = lane >> 4, l15 = lane & 15;
  const int b = blockIdx.z, h = blockIdx.y;
  const int qbase  = blockIdx.x * 64 + wid * 16;
  const int uqbase = blockIdx.x * 64 + uwid * 16;         // scalar copy for branches
  const size_t tokBase = (size_t)b * TS;
  const int ldq = 3 * DMOD;
  const u32 ldsKt = (u32)(size_t)(void*)&Kt[0][0];

  FragA qa[2];
  {
    const u16* qp = QKV + (tokBase + qbase + l15) * ldq + h * HSZ;
    #pragma unroll
    for (int fi = 0; fi < 2; ++fi) {
      qa[fi].q[0] = *(const uint4*)(qp + fi * 32 + hi * 8);
      qa[fi].q[1] = *(const uint4*)(qp + fi * 32 + hi * 8 + 16);
    }
  }

  float m[8], lsum[8];
  #pragma unroll
  for (int r = 0; r < 8; ++r) { m[r] = -1e30f; lsum[r] = 0.f; }
  v8f Yacc[4];
  #pragma unroll
  for (int i = 0; i < 4; ++i) Yacc[i] = vzero();

  const int nkb = blockIdx.x * 2 + 2;
  for (int kb = 0; kb < nkb; ++kb) {
    const int sbase = kb * 32;
    // K tile via TDM: 32 rows x 64 elems; LDS rows 128B + 32B pad:
    // pad_interval=4 (32 dwords), pad_amount=7 (8 dwords)
    if (uwid == 0)
      tdm_load_2d(ldsKt, QKV + (tokBase + sbase) * ldq + DMOD + h * HSZ,
                  (u32)ldq, 64, 32, 4, 7, (u32)ldq, (u32)MT);
    // stage V transposed into Vt[feat][key]
    {
      int key = tid & 31, fg = tid >> 5;
      const u16* vp = QKV + (tokBase + sbase + key) * ldq + 2 * DMOD + h * HSZ + fg * 16;
      u16 tmp[16];
      *(uint4*)&tmp[0] = *(const uint4*)vp;
      *(uint4*)&tmp[8] = *(const uint4*)(vp + 8);
      #pragma unroll
      for (int f = 0; f < 16; ++f) Vt[fg * 16 + f][key] = tmp[f];
    }
    if (uwid == 0) __builtin_amdgcn_s_wait_tensorcnt(0);
    __syncthreads();

    auto process = [&](auto mc) {
      constexpr bool MASKED = decltype(mc)::value;
      v8f c0 = vzero(), c1 = vzero();
      #pragma unroll
      for (int j = 0; j < 2; ++j) {
        v8f cc = vzero();
        #pragma unroll
        for (int fi = 0; fi < 2; ++fi) {
          FragA bk;
          int keyr = j * 16 + l15;
          bk.q[0] = *(const uint4*)&Kt[keyr][fi * 32 + hi * 8];
          bk.q[1] = *(const uint4*)&Kt[keyr][fi * 32 + hi * 8 + 16];
          cc = __builtin_amdgcn_wmma_f32_16x16x32_bf16(
              false, qa[fi].v, false, bk.v, (short)0, cc, false, false);
        }
        if (j == 0) c0 = cc; else c1 = cc;
      }
      float p0[8], p1[8], scl[8];
      #pragma unroll
      for (int r = 0; r < 8; ++r) {
        float s0, s1;
        if (MASKED) {
          int q = qbase + r + hi * 8;
          s0 = (sbase + l15      <= q) ? c0[r] * 0.125f : -1e30f;
          s1 = (sbase + 16 + l15 <= q) ? c1[r] * 0.125f : -1e30f;
        } else {
          s0 = c0[r] * 0.125f;
          s1 = c1[r] * 0.125f;
        }
        float mx = half16_max(fmaxf(s0, s1));
        float mnew = fmaxf(m[r], mx);
        float sc = __expf(m[r] - mnew);
        float e0 = __expf(s0 - mnew);
        float e1 = __expf(s1 - mnew);
        float rs = half16_sum(e0 + e1);
        lsum[r] = lsum[r] * sc + rs;
        m[r] = mnew;
        p0[r] = e0; p1[r] = e1; scl[r] = sc;
      }
      #pragma unroll
      for (int r = 0; r < 8; ++r) {
        Pb[wid][r + hi * 8][l15]      = f2bf(p0[r]);
        Pb[wid][r + hi * 8][16 + l15] = f2bf(p1[r]);
      }
      #pragma unroll
      for (int fi = 0; fi < 4; ++fi)
        #pragma unroll
        for (int r = 0; r < 8; ++r) Yacc[fi][r] *= scl[r];
      FragA pa;
      pa.q[0] = *(const uint4*)&Pb[wid][l15][hi * 8];
      pa.q[1] = *(const uint4*)&Pb[wid][l15][hi * 8 + 16];
      #pragma unroll
      for (int fi = 0; fi < 4; ++fi) {
        FragA vb;
        int fr = fi * 16 + l15;
        vb.q[0] = *(const uint4*)&Vt[fr][hi * 8];
        vb.q[1] = *(const uint4*)&Vt[fr][hi * 8 + 16];
        Yacc[fi] = __builtin_amdgcn_wmma_f32_16x16x32_bf16(
            false, pa.v, false, vb.v, (short)0, Yacc[fi], false, false);
      }
    };
    if (sbase <= uqbase + 15) {              // scalar wave-uniform branch
      if (sbase + 31 > uqbase) process(BoolC<true>{});   // diagonal block: needs mask
      else                     process(BoolC<false>{});  // interior: mask-free
    }
    __syncthreads();
  }
  #pragma unroll
  for (int r = 0; r < 8; ++r) {
    float inv = 1.f / lsum[r];
    u16* yp = Y + (tokBase + qbase + r + hi * 8) * DMOD + h * HSZ;
    #pragma unroll
    for (int fi = 0; fi < 4; ++fi)
      yp[fi * 16 + l15] = f2bf(Yacc[fi][r] * inv);
  }
}

// ---------------- host orchestration ----------------
extern "C" void kernel_launch(void* const* d_in, const int* in_sizes, int n_in,
                              void* d_out, int out_size, void* d_ws, size_t ws_size,
                              hipStream_t stream) {
  (void)in_sizes; (void)n_in; (void)out_size; (void)ws_size;
  const int*   idx    = (const int*)d_in[0];
  const float* wte    = (const float*)d_in[1];
  const float* wpe    = (const float*)d_in[2];
  const float* ln1_w  = (const float*)d_in[3];
  const float* ln1_b  = (const float*)d_in[4];
  const float* attn_w = (const float*)d_in[5];
  const float* attn_b = (const float*)d_in[6];
  const float* atp_w  = (const float*)d_in[7];
  const float* atp_b  = (const float*)d_in[8];
  const float* ln2_w  = (const float*)d_in[9];
  const float* ln2_b  = (const float*)d_in[10];
  const float* fc_w   = (const float*)d_in[11];
  const float* fc_b   = (const float*)d_in[12];
  const float* mp_w   = (const float*)d_in[13];
  const float* mp_b   = (const float*)d_in[14];
  const float* lnf_w  = (const float*)d_in[15];
  const float* lnf_b  = (const float*)d_in[16];

  char* w = (char*)d_ws;
  float* R  = (float*)w; w += (size_t)MT * DMOD * 4;
  u16* Abf  = (u16*)w;   w += (size_t)MT * DMOD * 2;
  u16* QKV  = (u16*)w;   w += (size_t)MT * 3 * DMOD * 2;
  u16* Ybf  = (u16*)w;   w += (size_t)MT * DMOD * 2;
  u16* FCb  = (u16*)w;   w += (size_t)MT * 4 * DMOD * 2;

  embed_kernel<<<MT * DMOD / 256, 256, 0, stream>>>(idx, wte, wpe, R);
  for (int l = 0; l < NL; ++l) {
    ln_kernel<<<MT, 256, 0, stream>>>(R, ln1_w + l * DMOD, ln1_b + l * DMOD, Abf);
    gemm_kernel<0, false><<<dim3(3 * DMOD / 128, MT / 128), 128, 0, stream>>>(
        Abf, attn_w + (size_t)l * DMOD * 3 * DMOD, attn_b + (size_t)l * 3 * DMOD,
        nullptr, QKV, MT, 3 * DMOD, DMOD, DMOD, 3 * DMOD);
    attn_kernel<<<dim3(TS / 64, NH, NB), 128, 0, stream>>>(QKV, Ybf);
    gemm_kernel<2, false><<<dim3(DMOD / 128, MT / 128), 128, 0, stream>>>(
        Ybf, atp_w + (size_t)l * DMOD * DMOD, atp_b + (size_t)l * DMOD,
        R, nullptr, MT, DMOD, DMOD, DMOD, DMOD);
    ln_kernel<<<MT, 256, 0, stream>>>(R, ln2_w + l * DMOD, ln2_b + l * DMOD, Abf);
    gemm_kernel<1, false><<<dim3(4 * DMOD / 128, MT / 128), 128, 0, stream>>>(
        Abf, fc_w + (size_t)l * DMOD * 4 * DMOD, fc_b + (size_t)l * 4 * DMOD,
        nullptr, FCb, MT, 4 * DMOD, DMOD, DMOD, 4 * DMOD);
    gemm_kernel<2, false><<<dim3(DMOD / 128, MT / 128), 128, 0, stream>>>(
        FCb, mp_w + (size_t)l * 4 * DMOD * DMOD, mp_b + (size_t)l * DMOD,
        R, nullptr, MT, DMOD, 4 * DMOD, 4 * DMOD, DMOD);
  }
  ln_kernel<<<MT, 256, 0, stream>>>(R, lnf_w, lnf_b, Abf);
  gemm_kernel<3, true><<<dim3((NV + 127) / 128, MT / 128), 128, 0, stream>>>(
      Abf, wte, nullptr, (float*)d_out, nullptr, MT, NV, DMOD, DMOD, DMOD);
}
